// MultiHeadMaskedAttention_4698694222413
// MI455X (gfx1250) — matrix operand order repaired and checked by reading, not compiled, not run on hardware
//
#include <hip/hip_runtime.h>
#include <hip/hip_bf16.h>
#include <stdint.h>

// ---------------- problem constants ----------------
#define BB 2
#define SS 2048
#define DD 1024
#define HH 16
#define HD 64
#define MM (BB * SS)          // 4096 rows
#define SCALE 0.125f          // 64^-0.5
#define LN_EPS 1e-12f

typedef __attribute__((ext_vector_type(16))) __bf16 v16bf;
typedef __attribute__((ext_vector_type(8)))  float  v8f;
typedef __attribute__((ext_vector_type(4)))  unsigned int v4u;
typedef __attribute__((ext_vector_type(8)))  int v8i;
typedef __attribute__((ext_vector_type(4)))  int v4i;

// fp32 -> bf16 (round to nearest even), raw u16
__device__ __forceinline__ unsigned short f2bf(float f) {
    union { float f; unsigned u; } v; v.f = f;
    unsigned r = v.u + 0x7fffu + ((v.u >> 16) & 1u);
    return (unsigned short)(r >> 16);
}

// ---- Tensor Data Mover: 2D tile (tile_k x tile_rows) global->LDS, bf16 ----
__device__ __forceinline__ void tdm_load_2d(unsigned short* lds,
                                            const unsigned short* gptr,
                                            unsigned tile_k, unsigned tile_rows,
                                            unsigned tensor_k, unsigned tensor_rows,
                                            unsigned row_stride) {
    unsigned lds_addr = (unsigned)(unsigned long long)(void*)lds;
    unsigned long long ga = (unsigned long long)gptr;
    v4u g0 = { 1u,                                   // count = 1 valid descriptor
               lds_addr,                              // lds_addr [63:32]
               (unsigned)(ga & 0xffffffffu),          // global_addr lo
               (unsigned)((ga >> 32) & 0x01ffffffu) | (2u << 30) };  // addr hi | type=2
    v8i g1;
    g1[0] = (int)(1u << 16);                                           // data_size = 1 (2B)
    g1[1] = (int)((tensor_k & 0xffffu) << 16);                         // tensor_dim0 lo16
    g1[2] = (int)((tensor_k >> 16) | ((tensor_rows & 0xffffu) << 16)); // dim0 hi | dim1 lo
    g1[3] = (int)((tensor_rows >> 16) | (tile_k << 16));               // dim1 hi | tile_dim0
    g1[4] = (int)tile_rows;                                            // tile_dim1 (dim2=0)
    g1[5] = (int)row_stride;                                           // tensor_dim0_stride lo32
    g1[6] = 0;
    g1[7] = 0;
    v4i gz = { 0, 0, 0, 0 };
#if defined(__clang_major__) && (__clang_major__ >= 23)
    v8i gz8 = { 0, 0, 0, 0, 0, 0, 0, 0 };
    __builtin_amdgcn_tensor_load_to_lds(g0, g1, gz, gz, gz8, 0);
#else
    __builtin_amdgcn_tensor_load_to_lds(g0, g1, gz, gz, 0);
#endif
}

// ---- WMMA fragment gathers from LDS (bf16 as u16), wave32 layouts ----
__device__ __forceinline__ v16bf ld_fragA(const unsigned short* p, int ld, int lane) {
    int m = lane & 15, hi = lane >> 4;
    v16bf a;
    ((uint4*)&a)[0] = *(const uint4*)(p + m * ld + hi * 8);
    ((uint4*)&a)[1] = *(const uint4*)(p + m * ld + 16 + hi * 8);
    return a;
}
__device__ __forceinline__ v16bf ld_fragB(const unsigned short* p, int ld, int lane) {
    int n = lane & 15, hi = lane >> 4;
    v16bf b;
    ((uint4*)&b)[0] = *(const uint4*)(p + n * ld + hi * 8);
    ((uint4*)&b)[1] = *(const uint4*)(p + n * ld + 16 + hi * 8);
    return b;
}

__device__ __forceinline__ float red_max16(float x) {
#pragma unroll
    for (int m = 1; m < 16; m <<= 1) x = fmaxf(x, __shfl_xor(x, m, 32));
    return x;
}
__device__ __forceinline__ float red_sum16(float x) {
#pragma unroll
    for (int m = 1; m < 16; m <<= 1) x += __shfl_xor(x, m, 32);
    return x;
}

// ---------------- one-shot fp32 -> bf16 conversion (8 elems/thread) ----------------
__global__ __launch_bounds__(256) void cvt_f32_bf16(const float* __restrict__ src,
                                                    unsigned short* __restrict__ dst, int n) {
    int i = (blockIdx.x * 256 + threadIdx.x) * 8;
    if (i + 8 > n) return;
    float4 a = ((const float4*)(src + i))[0];
    float4 b = ((const float4*)(src + i))[1];
    unsigned short t[8] = { f2bf(a.x), f2bf(a.y), f2bf(a.z), f2bf(a.w),
                            f2bf(b.x), f2bf(b.y), f2bf(b.z), f2bf(b.w) };
    uint4 u; __builtin_memcpy(&u, t, 16);
    *(uint4*)(dst + i) = u;
}

// ---------------- GEMM: Y[M,N] = A[M,K](bf16) @ W[N,K](bf16)^T + bias (+resid) ----
// Block tile 64x128, 8 waves each computing 16x64 (1 A-frag -> 4 WMMAs per k-step).
// Double-buffered TDM staging overlaps DMA with WMMA compute.
__global__ __launch_bounds__(256) void gemm_bf16_wmma(
    const unsigned short* __restrict__ A, const unsigned short* __restrict__ W,
    const float* __restrict__ bias, const float* __restrict__ resid,
    float* __restrict__ outF, unsigned short* __restrict__ outB,
    int Mn, int Nn, int Kn)
{
    __shared__ __align__(16) unsigned short As[2][64 * 32];
    __shared__ __align__(16) unsigned short Bs[2][128 * 32];
    const int tid = threadIdx.x;
    const int lane = tid & 31, w = tid >> 5;
    const int wm = w & 3, wn = w >> 2;          // 4 x 2 waves -> 64 rows x 128 cols
    const int m0 = blockIdx.y * 64, n0 = blockIdx.x * 128;

    v8f acc[4];
#pragma unroll
    for (int j = 0; j < 4; ++j)
#pragma unroll
        for (int i = 0; i < 8; ++i) acc[j][i] = 0.f;

    if (w == 0) {   // prologue: DMA first k-tile into buffer 0
        tdm_load_2d(As[0], A + (size_t)m0 * Kn, 32, 64, Kn, Mn, Kn);
        tdm_load_2d(Bs[0], W + (size_t)n0 * Kn, 32, 128, Kn, Nn, Kn);
    }

    int cur = 0;
    for (int k0 = 0; k0 < Kn; k0 += 32) {
        if (w == 0) {
            if (k0 + 32 < Kn) {   // DMA next k-tile into the other buffer, then
                tdm_load_2d(As[cur ^ 1], A + (size_t)m0 * Kn + k0 + 32, 32, 64, Kn, Mn, Kn);
                tdm_load_2d(Bs[cur ^ 1], W + (size_t)n0 * Kn + k0 + 32, 32, 128, Kn, Nn, Kn);
                __builtin_amdgcn_s_wait_tensorcnt(2);   // in-order: current pair landed
            } else {
                __builtin_amdgcn_s_wait_tensorcnt(0);
            }
        }
        __syncthreads();
        v16bf fa = ld_fragA(As[cur] + wm * 16 * 32, 32, lane);
#pragma unroll
        for (int j = 0; j < 4; ++j) {
            v16bf fb = ld_fragB(Bs[cur] + (wn * 64 + j * 16) * 32, 32, lane);
            acc[j] = __builtin_amdgcn_wmma_f32_16x16x32_bf16(false, fa, false, fb,
                                                             (short)0, acc[j], false, false);
        }
        __syncthreads();
        cur ^= 1;
    }

    const int nl = lane & 15, hi = lane >> 4;
#pragma unroll
    for (int j = 0; j < 4; ++j) {
        const int col = n0 + wn * 64 + j * 16 + nl;
        const float bv = bias ? bias[col] : 0.f;
#pragma unroll
        for (int r = 0; r < 8; ++r) {
            int row = m0 + wm * 16 + r + hi * 8;
            size_t idx = (size_t)row * Nn + col;
            float y = acc[j][r] + bv;
            if (resid) y += resid[idx];
            if (outF)  outF[idx] = y;
            if (outB)  outB[idx] = f2bf(y);
        }
    }
}

// ---------------- Flash attention: one wave per (b,h,16-query tile) ----------------
__global__ __launch_bounds__(32) void flash_attn_wmma(
    const unsigned short* __restrict__ Q, const unsigned short* __restrict__ K,
    const unsigned short* __restrict__ V, unsigned short* __restrict__ O)
{
    __shared__ __align__(16) unsigned short qs[16 * 64];      // [q][d]
    __shared__ __align__(16) unsigned short ks[2][32 * 64];   // [key][d], double buffered
    __shared__ __align__(16) unsigned short vs[64 * 32];      // transposed [d][key]
    __shared__ __align__(16) unsigned short ps[16 * 32];      // bf16 probs [q][key]

    const int lane = threadIdx.x;
    const int QT = SS / 16;
    int bid = blockIdx.x;
    const int qt = bid % QT; bid /= QT;
    const int h  = bid % HH;
    const int b  = bid / HH;
    const size_t rowBase = (size_t)b * SS;
    const size_t headCol = (size_t)h * HD;
    const int q0 = qt * 16;
    const int hi = lane >> 4, nl = lane & 15;
    const int nkb = ((q0 + 15) >> 5) + 1;       // causal: key blocks of 32

    // prologue DMAs: Q tile, then first K tile (TDM completes in order)
    tdm_load_2d(qs, Q + (rowBase + q0) * DD + headCol, 64, 16, DD, MM, DD);
    tdm_load_2d(ks[0], K + rowBase * DD + headCol, 64, 32, DD, MM, DD);

    float mrun[8], lrun[8], corr[8];
    v8f acc[4];
#pragma unroll
    for (int r = 0; r < 8; ++r) { mrun[r] = -INFINITY; lrun[r] = 0.f; }
#pragma unroll
    for (int dt = 0; dt < 4; ++dt)
#pragma unroll
        for (int r = 0; r < 8; ++r) acc[dt][r] = 0.f;

    int cur = 0;
    for (int kb = 0; kb < nkb; ++kb) {
        const int kbase = kb * 32;
        if (kb + 1 < nkb)   // DMA next K tile into alternate buffer
            tdm_load_2d(ks[cur ^ 1], K + (rowBase + kbase + 32) * DD + headCol, 64, 32, DD, MM, DD);

        // stage V block transposed (overlaps in-flight K DMA) -> vs[d][key]
#pragma unroll
        for (int i = 0; i < 8; ++i) {
            int c = lane + 32 * i;
            int kk = c >> 3, d0 = (c & 7) * 8;
            const unsigned short* src = V + (rowBase + kbase + kk) * DD + headCol + d0;
            uint4 v4 = *(const uint4*)src;
            unsigned short hsv[8]; __builtin_memcpy(hsv, &v4, 16);
#pragma unroll
            for (int j = 0; j < 8; ++j) vs[(d0 + j) * 32 + kk] = hsv[j];
        }
        if (kb + 1 < nkb) __builtin_amdgcn_s_wait_tensorcnt(1);  // current K landed
        else              __builtin_amdgcn_s_wait_tensorcnt(0);
        __syncthreads();

        // scores: two 16x16 tiles over 32 keys, K-dim = HD = 64 (2 wmma steps each)
        v8f s[2];
#pragma unroll
        for (int t = 0; t < 2; ++t) {
            v8f c8;
#pragma unroll
            for (int r = 0; r < 8; ++r) c8[r] = 0.f;
            c8 = __builtin_amdgcn_wmma_f32_16x16x32_bf16(false, ld_fragA(qs, 64, lane),
                    false, ld_fragB(ks[cur] + t * 16 * 64, 64, lane), (short)0, c8, false, false);
            c8 = __builtin_amdgcn_wmma_f32_16x16x32_bf16(false, ld_fragA(qs + 32, 64, lane),
                    false, ld_fragB(ks[cur] + t * 16 * 64 + 32, 64, lane), (short)0, c8, false, false);
            const int key = kbase + t * 16 + nl;
#pragma unroll
            for (int r = 0; r < 8; ++r) {
                int qrow = q0 + r + hi * 8;
                c8[r] = (key > qrow) ? -INFINITY : c8[r] * SCALE;
            }
            s[t] = c8;
        }

        // online softmax (rows replicated across 16 lanes)
#pragma unroll
        for (int r = 0; r < 8; ++r) {
            float mx = red_max16(fmaxf(s[0][r], s[1][r]));
            float nm = fmaxf(mrun[r], mx);
            float cr = __expf(mrun[r] - nm);
            float p0 = __expf(s[0][r] - nm);
            float p1 = __expf(s[1][r] - nm);
            float pz = red_sum16(p0 + p1);
            lrun[r] = lrun[r] * cr + pz;
            mrun[r] = nm;
            corr[r] = cr;
            ps[(r + hi * 8) * 32 + nl]      = f2bf(p0);
            ps[(r + hi * 8) * 32 + 16 + nl] = f2bf(p1);
        }
        __syncthreads();

        // O = O*corr + P(16x32) @ V(32x64), 4 d-tiles
        v16bf fp = ld_fragA(ps, 32, lane);
#pragma unroll
        for (int dt = 0; dt < 4; ++dt) {
#pragma unroll
            for (int r = 0; r < 8; ++r) acc[dt][r] *= corr[r];
            v16bf fv = ld_fragB(vs + dt * 16 * 32, 32, lane);
            acc[dt] = __builtin_amdgcn_wmma_f32_16x16x32_bf16(false, fp, false, fv,
                                                              (short)0, acc[dt], false, false);
        }
        __syncthreads();
        cur ^= 1;
    }

    // normalize + write bf16 O (consumed by the output-projection GEMM)
#pragma unroll
    for (int dt = 0; dt < 4; ++dt)
#pragma unroll
        for (int r = 0; r < 8; ++r) {
            int row = q0 + r + hi * 8;
            size_t col = headCol + dt * 16 + nl;
            O[(rowBase + row) * DD + col] = f2bf(acc[dt][r] / lrun[r]);
        }
}

// ---------------- LayerNorm over D=1024 per row ----------------
__global__ __launch_bounds__(256) void layernorm_k(
    const float* __restrict__ X, const float* __restrict__ gamma,
    const float* __restrict__ beta, float* __restrict__ out)
{
    __shared__ float s1[256], s2[256];
    const int row = blockIdx.x, tid = threadIdx.x;
    const float* x = X + (size_t)row * DD;
    float a = 0.f, b = 0.f, vals[4];
#pragma unroll
    for (int i = 0; i < 4; ++i) {
        float v = x[tid + 256 * i];
        vals[i] = v; a += v; b += v * v;
    }
    s1[tid] = a; s2[tid] = b;
    __syncthreads();
    for (int st = 128; st > 0; st >>= 1) {
        if (tid < st) { s1[tid] += s1[tid + st]; s2[tid] += s2[tid + st]; }
        __syncthreads();
    }
    float mean = s1[0] * (1.f / DD);
    float var  = s2[0] * (1.f / DD) - mean * mean;
    float rs   = rsqrtf(var + LN_EPS);
#pragma unroll
    for (int i = 0; i < 4; ++i) {
        int c = tid + 256 * i;
        out[(size_t)row * DD + c] = (vals[i] - mean) * rs * gamma[c] + beta[c];
    }
}

// ---------------- host launcher ----------------
extern "C" void kernel_launch(void* const* d_in, const int* in_sizes, int n_in,
                              void* d_out, int out_size, void* d_ws, size_t ws_size,
                              hipStream_t stream) {
    const float* queries = (const float*)d_in[0];
    const float* keys    = (const float*)d_in[1];
    const float* values  = (const float*)d_in[2];
    // d_in[3] = causal ignore mask (implemented analytically)
    const float* Wq = (const float*)d_in[4];  const float* bq = (const float*)d_in[5];
    const float* Wk = (const float*)d_in[6];  const float* bk = (const float*)d_in[7];
    const float* Wv = (const float*)d_in[8];  const float* bv = (const float*)d_in[9];
    const float* Wo = (const float*)d_in[10]; const float* bo = (const float*)d_in[11];
    const float* ln_g = (const float*)d_in[12];
    const float* ln_b = (const float*)d_in[13];
    float* out = (float*)d_out;

    const size_t act = (size_t)MM * DD;      // 4M activation elems
    const size_t wgt = (size_t)DD * DD;      // 1M weight elems
    char* ws = (char*)d_ws;
    size_t off = 0;
    unsigned short* xq16 = (unsigned short*)(ws + off); off += act * 2;
    unsigned short* xk16 = (unsigned short*)(ws + off); off += act * 2;
    unsigned short* xv16 = (unsigned short*)(ws + off); off += act * 2;
    unsigned short* wq16 = (unsigned short*)(ws + off); off += wgt * 2;
    unsigned short* wk16 = (unsigned short*)(ws + off); off += wgt * 2;
    unsigned short* wv16 = (unsigned short*)(ws + off); off += wgt * 2;
    unsigned short* wo16 = (unsigned short*)(ws + off); off += wgt * 2;
    unsigned short* q16  = (unsigned short*)(ws + off); off += act * 2;
    unsigned short* k16  = (unsigned short*)(ws + off); off += act * 2;
    unsigned short* v16  = (unsigned short*)(ws + off); off += act * 2;
    unsigned short* o16  = (unsigned short*)(ws + off); off += act * 2;
    float* xres = (float*)(ws + off); off += act * 4;
    (void)ws_size; (void)in_sizes; (void)n_in; (void)out_size;

    // one-shot fp32 -> bf16 of activations and weights
    cvt_f32_bf16<<<dim3(act / 2048), 256, 0, stream>>>(queries, xq16, (int)act);
    cvt_f32_bf16<<<dim3(act / 2048), 256, 0, stream>>>(keys,    xk16, (int)act);
    cvt_f32_bf16<<<dim3(act / 2048), 256, 0, stream>>>(values,  xv16, (int)act);
    cvt_f32_bf16<<<dim3(wgt / 2048), 256, 0, stream>>>(Wq, wq16, (int)wgt);
    cvt_f32_bf16<<<dim3(wgt / 2048), 256, 0, stream>>>(Wk, wk16, (int)wgt);
    cvt_f32_bf16<<<dim3(wgt / 2048), 256, 0, stream>>>(Wv, wv16, (int)wgt);
    cvt_f32_bf16<<<dim3(wgt / 2048), 256, 0, stream>>>(Wo, wo16, (int)wgt);

    dim3 ggrid(DD / 128, MM / 64);   // (8, 64)
    gemm_bf16_wmma<<<ggrid, 256, 0, stream>>>(xq16, wq16, bq, nullptr, nullptr, q16, MM, DD, DD);
    gemm_bf16_wmma<<<ggrid, 256, 0, stream>>>(xk16, wk16, bk, nullptr, nullptr, k16, MM, DD, DD);
    gemm_bf16_wmma<<<ggrid, 256, 0, stream>>>(xv16, wv16, bv, nullptr, nullptr, v16, MM, DD, DD);
    // causal flash attention -> bf16 O
    flash_attn_wmma<<<dim3(BB * HH * (SS / 16)), 32, 0, stream>>>(q16, k16, v16, o16);
    // output projection + residual(values fp32)
    gemm_bf16_wmma<<<ggrid, 256, 0, stream>>>(o16, wo16, bo, values, xres, nullptr, MM, DD, DD);
    // LayerNorm
    layernorm_k<<<dim3(MM), 256, 0, stream>>>(xres, ln_g, ln_b, out);
}
